// Gemma2QuantizedMLP_71700184040135
// MI455X (gfx1250) — compile-verified, matrix operand order checked by reading, not run on hardware
//
#include <hip/hip_runtime.h>
#include <hip/hip_bf16.h>
#include <math.h>

typedef __bf16 bf16;
typedef __attribute__((ext_vector_type(16))) __bf16 v16bf;
typedef __attribute__((ext_vector_type(8)))  __bf16 v8bf;
typedef __attribute__((ext_vector_type(4)))  __bf16 v4bf;
typedef __attribute__((ext_vector_type(8)))  float  v8f;
typedef __attribute__((ext_vector_type(4)))  float  v4f;
typedef __attribute__((ext_vector_type(4)))  int    v4i;

#define DIM_H 3584
#define DIM_I 14336
#define DIM_M 8192     // B*S = 4*2048
#define QGRP  128
#define XS_STR 40      // bf16 LDS row stride (80B)
#define WQ_STR 36      // int32 LDS row stride (144B, 16B-aligned rows)

// ---- CDNA5 async global->LDS copy (ASYNCcnt path), 16B per lane ----
__device__ inline void async_copy16(void* dst_lds, const void* src) {
  unsigned lds = (unsigned)(unsigned long long)dst_lds;   // low 32b = LDS offset
  asm volatile("global_load_async_to_lds_b128 %0, %1, off"
               :: "v"(lds), "v"((unsigned long long)src) : "memory");
}
__device__ inline void wait_async0() {
  asm volatile("s_wait_asynccnt 0x0" ::: "memory");
}

// Build v16bf WMMA fragment from two contiguous 16B bf16 LDS runs
// (ISA 16-bit layout: lanes 0-15 K={0..7,16..23}, lanes 16-31 K={8..15,24..31})
__device__ inline v16bf frag16(const bf16* p) {
  v8bf lo = *(const v8bf*)(p);
  v8bf hi = *(const v8bf*)(p + 16);
  return __builtin_shufflevector(lo, hi, 0,1,2,3,4,5,6,7,8,9,10,11,12,13,14,15);
}

// Dequantize one B fragment from raw int32 LDS tile (16 elems) with one scale
__device__ inline v16bf dequant_frag(const int* p, float s) {
  v4i q0 = *(const v4i*)(p);
  v4i q1 = *(const v4i*)(p + 4);
  v4i q2 = *(const v4i*)(p + 16);
  v4i q3 = *(const v4i*)(p + 20);
  v16bf b;
#pragma unroll
  for (int j = 0; j < 4; ++j) {
    b[j]      = (bf16)((float)q0[j] * s);
    b[4 + j]  = (bf16)((float)q1[j] * s);
    b[8 + j]  = (bf16)((float)q2[j] * s);
    b[12 + j] = (bf16)((float)q3[j] * s);
  }
  return b;
}

__device__ inline float gelu_tanh(float g) {
  float t = 0.7978845608028654f * (g + 0.044715f * g * g * g);
  return 0.5f * g * (1.0f + tanhf(t));
}

// ---------------------------------------------------------------------------
// Pass 0: x fp32 -> bf16
// ---------------------------------------------------------------------------
__global__ __launch_bounds__(256) void cvt_x_bf16(const float* __restrict__ x,
                                                  bf16* __restrict__ xb, int n) {
  int i = (blockIdx.x * 256 + threadIdx.x) * 4;
  if (i < n) {
    v4f v = *(const v4f*)(x + i);
    v4bf o;
    o[0] = (bf16)v[0]; o[1] = (bf16)v[1]; o[2] = (bf16)v[2]; o[3] = (bf16)v[3];
    *(v4bf*)(xb + i) = o;
  }
}

// ---------------------------------------------------------------------------
// Pass 1: h = gelu_tanh(x @ Wg^T) * (x @ Wu^T); async-staged, double-buffered
// Block tile 64(M) x 128(N); 8 waves = 2x4; each wave 32x32 per matrix.
// ---------------------------------------------------------------------------
__global__ __launch_bounds__(256) void gateup_wmma(
    const bf16* __restrict__ xb,
    const int*  __restrict__ gq, const float* __restrict__ gs,
    const int*  __restrict__ uq, const float* __restrict__ us,
    bf16* __restrict__ hout) {
  __shared__ bf16 xs[2][64 * XS_STR];
  __shared__ int  wq[2][2][128 * WQ_STR];   // [buf][gate/up]

  const int t    = threadIdx.x;
  const int m0   = blockIdx.y * 64;
  const int n0   = blockIdx.x * 128;
  const int wave = t >> 5;
  const int lane = t & 31;
  const int wm   = wave & 1;
  const int wn   = wave >> 1;
  const int half = lane >> 4;
  const int ln   = lane & 15;
  const int c0   = half * 8;

  // staging assignments (16B granules)
  const int xr = t >> 2, xc = (t & 3) * 8;         // x: 64 rows, 4 thr/row
  const int wr = t >> 1, wc = (t & 1) * 16;        // w: 128 rows, 2 thr/row, 4x16B
  const bf16* xg  = xb + (size_t)(m0 + xr) * DIM_H + xc;
  const int*  gqg = gq + (size_t)(n0 + wr) * DIM_H + wc;
  const int*  uqg = uq + (size_t)(n0 + wr) * DIM_H + wc;

  // per-wave scale row bases (row depends on ni)
  const int nrow0 = n0 + wn * 32 + ln;             // ni=0 row
  const float* gs0 = gs + (size_t)(nrow0)      * (DIM_H / QGRP);
  const float* gs1 = gs + (size_t)(nrow0 + 16) * (DIM_H / QGRP);
  const float* us0 = us + (size_t)(nrow0)      * (DIM_H / QGRP);
  const float* us1 = us + (size_t)(nrow0 + 16) * (DIM_H / QGRP);

  v8f ag[2][2], au[2][2];
#pragma unroll
  for (int i = 0; i < 2; ++i)
#pragma unroll
    for (int j = 0; j < 2; ++j) { ag[i][j] = (v8f){}; au[i][j] = (v8f){}; }

  auto stage = [&](int k0, int buf) {
    async_copy16(&xs[buf][xr * XS_STR + xc], xg + k0);
#pragma unroll
    for (int j = 0; j < 4; ++j) {
      async_copy16(&wq[buf][0][wr * WQ_STR + wc + j * 4], gqg + k0 + j * 4);
      async_copy16(&wq[buf][1][wr * WQ_STR + wc + j * 4], uqg + k0 + j * 4);
    }
  };

  int cur = 0;
  stage(0, 0);
  wait_async0();
  __syncthreads();

  for (int g = 0; g < DIM_H / QGRP; ++g) {
    const float sg[2] = { gs0[g], gs1[g] };
    const float su[2] = { us0[g], us1[g] };
#pragma unroll
    for (int kk = 0; kk < 4; ++kk) {
      const int k0  = g * QGRP + kk * 32;
      const int nxt = cur ^ 1;
      if (k0 + 32 < DIM_H) stage(k0 + 32, nxt);   // async prefetch next slab

      v16bf a[2], bg[2], bu[2];
#pragma unroll
      for (int mi = 0; mi < 2; ++mi)
        a[mi] = frag16(&xs[cur][(wm * 32 + mi * 16 + ln) * XS_STR + c0]);
#pragma unroll
      for (int ni = 0; ni < 2; ++ni) {
        const int roff = (wn * 32 + ni * 16 + ln) * WQ_STR + c0;
        bg[ni] = dequant_frag(&wq[cur][0][roff], sg[ni]);
        bu[ni] = dequant_frag(&wq[cur][1][roff], su[ni]);
      }
#pragma unroll
      for (int mi = 0; mi < 2; ++mi)
#pragma unroll
        for (int ni = 0; ni < 2; ++ni) {
          ag[mi][ni] = __builtin_amdgcn_wmma_f32_16x16x32_bf16(
              false, a[mi], false, bg[ni], (short)0, ag[mi][ni], false, false);
          au[mi][ni] = __builtin_amdgcn_wmma_f32_16x16x32_bf16(
              false, a[mi], false, bu[ni], (short)0, au[mi][ni], false, false);
        }
      wait_async0();
      __syncthreads();
      cur = nxt;
    }
  }

  const int row0 = m0 + wm * 32;
  const int col0 = n0 + wn * 32;
#pragma unroll
  for (int mi = 0; mi < 2; ++mi)
#pragma unroll
    for (int ni = 0; ni < 2; ++ni)
#pragma unroll
      for (int e = 0; e < 8; ++e) {
        float hv = gelu_tanh(ag[mi][ni][e]) * au[mi][ni][e];
        int r = row0 + mi * 16 + half * 8 + e;
        int c = col0 + ni * 16 + ln;
        hout[(size_t)r * DIM_I + c] = (bf16)hv;
      }
}

// ---------------------------------------------------------------------------
// Pass 2: out = h @ Wd^T, K = 14336, N = 3584, fp32 out
// ---------------------------------------------------------------------------
__global__ __launch_bounds__(256) void down_wmma(
    const bf16* __restrict__ h,
    const int*  __restrict__ dq, const float* __restrict__ ds,
    float* __restrict__ out) {
  __shared__ bf16 hs[2][64 * XS_STR];
  __shared__ int  wd[2][128 * WQ_STR];

  const int t    = threadIdx.x;
  const int m0   = blockIdx.y * 64;
  const int n0   = blockIdx.x * 128;
  const int wave = t >> 5;
  const int lane = t & 31;
  const int wm   = wave & 1;
  const int wn   = wave >> 1;
  const int half = lane >> 4;
  const int ln   = lane & 15;
  const int c0   = half * 8;

  const int xr = t >> 2, xc = (t & 3) * 8;
  const int wr = t >> 1, wc = (t & 1) * 16;
  const bf16* hg  = h  + (size_t)(m0 + xr) * DIM_I + xc;
  const int*  dqg = dq + (size_t)(n0 + wr) * DIM_I + wc;

  const int nrow0 = n0 + wn * 32 + ln;
  const float* ds0 = ds + (size_t)(nrow0)      * (DIM_I / QGRP);
  const float* ds1 = ds + (size_t)(nrow0 + 16) * (DIM_I / QGRP);

  v8f acc[2][2];
#pragma unroll
  for (int i = 0; i < 2; ++i)
#pragma unroll
    for (int j = 0; j < 2; ++j) acc[i][j] = (v8f){};

  auto stage = [&](int k0, int buf) {
    async_copy16(&hs[buf][xr * XS_STR + xc], hg + k0);
#pragma unroll
    for (int j = 0; j < 4; ++j)
      async_copy16(&wd[buf][wr * WQ_STR + wc + j * 4], dqg + k0 + j * 4);
  };

  int cur = 0;
  stage(0, 0);
  wait_async0();
  __syncthreads();

  for (int g = 0; g < DIM_I / QGRP; ++g) {
    const float sd[2] = { ds0[g], ds1[g] };
#pragma unroll
    for (int kk = 0; kk < 4; ++kk) {
      const int k0  = g * QGRP + kk * 32;
      const int nxt = cur ^ 1;
      if (k0 + 32 < DIM_I) stage(k0 + 32, nxt);

      v16bf a[2], b[2];
#pragma unroll
      for (int mi = 0; mi < 2; ++mi)
        a[mi] = frag16(&hs[cur][(wm * 32 + mi * 16 + ln) * XS_STR + c0]);
#pragma unroll
      for (int ni = 0; ni < 2; ++ni)
        b[ni] = dequant_frag(&wd[cur][(wn * 32 + ni * 16 + ln) * WQ_STR + c0], sd[ni]);
#pragma unroll
      for (int mi = 0; mi < 2; ++mi)
#pragma unroll
        for (int ni = 0; ni < 2; ++ni)
          acc[mi][ni] = __builtin_amdgcn_wmma_f32_16x16x32_bf16(
              false, a[mi], false, b[ni], (short)0, acc[mi][ni], false, false);
      wait_async0();
      __syncthreads();
      cur = nxt;
    }
  }

  const int row0 = m0 + wm * 32;
  const int col0 = n0 + wn * 32;
#pragma unroll
  for (int mi = 0; mi < 2; ++mi)
#pragma unroll
    for (int ni = 0; ni < 2; ++ni)
#pragma unroll
      for (int e = 0; e < 8; ++e) {
        int r = row0 + mi * 16 + half * 8 + e;
        int c = col0 + ni * 16 + ln;
        out[(size_t)r * DIM_H + c] = acc[mi][ni][e];
      }
}

extern "C" void kernel_launch(void* const* d_in, const int* in_sizes, int n_in,
                              void* d_out, int out_size, void* d_ws, size_t ws_size,
                              hipStream_t stream) {
  const float* x   = (const float*)d_in[0];
  const int*   gq  = (const int*)  d_in[1];
  const float* gs  = (const float*)d_in[2];
  const int*   uq  = (const int*)  d_in[3];
  const float* us  = (const float*)d_in[4];
  const int*   dq  = (const int*)  d_in[5];
  const float* dsc = (const float*)d_in[6];
  float*       out = (float*)d_out;

  bf16* xb = (bf16*)d_ws;                                        // 58.7 MB
  bf16* hb = (bf16*)((char*)d_ws + (size_t)DIM_M * DIM_H * 2);   // 235 MB

  cvt_x_bf16<<<(DIM_M * DIM_H) / 1024, 256, 0, stream>>>(x, xb, DIM_M * DIM_H);
  gateup_wmma<<<dim3(DIM_I / 128, DIM_M / 64), 256, 0, stream>>>(xb, gq, gs, uq, us, hb);
  down_wmma<<<dim3(DIM_H / 128, DIM_M / 64), 256, 0, stream>>>(hb, dq, dsc, out);
}
